// ConcatAttentionCoverage_15479062135445
// MI455X (gfx1250) — matrix operand order repaired and checked by reading, not compiled, not run on hardware
//
#include <hip/hip_runtime.h>
#include <hip/hip_bf16.h>
#include <math.h>

// MI455X / gfx1250, wave32.
// Dominant GEMM (context @ W_pre^T: 65536 x 1024 x 1024, 137 GFLOP) runs as a
// bf16x3 split-GEMM on v_wmma_f32_16x16x32_bf16: a = a_hi + a_lo (bf16),
// acc += ah*bh + ah*bl + al*bh (fp32 accumulate). 3 WMMA issues per K=32 vs 8
// for the f32 pipe (K=4/issue) -> ~2.7x on the compute-bound kernel, with
// ~1e-4 relative error (lo*lo term ~2^-16 dropped). HBM floor is ~33us
// (768 MB @ 23.3 TB/s); this moves compute much closer to it.

typedef float  v8f  __attribute__((ext_vector_type(8)));
typedef __bf16 v4bf __attribute__((ext_vector_type(4)));
typedef __bf16 v8bf __attribute__((ext_vector_type(8)));
typedef __bf16 v16bf __attribute__((ext_vector_type(16)));

#define NB   64      // batch
#define NS   1024    // sequence
#define ND   1024    // att_dim == attend_dim == query_dim
#define MTOT (NB * NS)

// ---------------------------------------------------------------------------
// target[b][d] = sum_q input[b][q] * W_q[d][q]   (0.1% of the FLOPs; fp32)
// 64 consecutive threads share d -> W_q row is wave-uniform.
// ---------------------------------------------------------------------------
__global__ __launch_bounds__(256)
void target_kernel(const float* __restrict__ input,
                   const float* __restrict__ Wq,
                   float* __restrict__ target) {
  int idx = blockIdx.x * 256 + threadIdx.x;      // 65536 threads
  int b = idx & (NB - 1);
  int d = idx >> 6;
  const float4* in4 = (const float4*)(input + (size_t)b * ND);
  const float4* wq4 = (const float4*)(Wq + (size_t)d * ND);
  float acc = 0.f;
#pragma unroll 8
  for (int q = 0; q < ND / 4; ++q) {
    float4 a = in4[q], w = wq4[q];
    acc += a.x * w.x + a.y * w.y + a.z * w.z + a.w * w.w;
  }
  target[b * ND + d] = acc;
}

// ---------------------------------------------------------------------------
// Fused: precompute = context @ W_pre^T + b_pre ; tmp = tanh(precompute +
// target + cov*W_cov) ; energy[m] += tmp . W_v  (atomic partial per N-tile)
//
// Block: 256 threads = 8 waves. Tile 128(M) x 64(N), K chunks of 32 in
// double-buffered LDS (bf16 hi/lo pairs). Wave w owns rows [16w,16w+16) and
// all 64 columns (4 x v8f accumulators).
// ---------------------------------------------------------------------------
#define TM  128
#define TN  64
#define KC  32
#define LDW 40               // padded LDS row stride in halves (80B rows)
#define NCHUNK (ND / KC)     // 32

// Split a float4 into bf16 hi (RNE) + bf16 lo (residual) vectors.
__device__ __forceinline__ void split4(const float4 v, v4bf& h, v4bf& l) {
  const __bf16 h0 = (__bf16)v.x, h1 = (__bf16)v.y;
  const __bf16 h2 = (__bf16)v.z, h3 = (__bf16)v.w;
  h = (v4bf){h0, h1, h2, h3};
  l = (v4bf){(__bf16)(v.x - (float)h0), (__bf16)(v.y - (float)h1),
             (__bf16)(v.z - (float)h2), (__bf16)(v.w - (float)h3)};
}

__global__ __launch_bounds__(256)
void fused_gemm_tanh_energy(const float* __restrict__ context,   // (B*S, ND)
                            const float* __restrict__ Wpre,      // (ND, ND)
                            const float* __restrict__ bpre,      // (ND)
                            const float* __restrict__ target,    // (B, ND)
                            const float* __restrict__ coverage,  // (B*S)
                            const float* __restrict__ Wv,        // (ND)
                            const float* __restrict__ Wcov,      // (ND)
                            float* __restrict__ precompute_out,  // (B*S, ND)
                            float* __restrict__ energy) {        // (B*S)
  // 2 * (128*40 + 128*40 + 64*40 + 64*40) halves = 61440 B
  __shared__ __bf16 Ah[2][TM * LDW];
  __shared__ __bf16 Al[2][TM * LDW];
  __shared__ __bf16 Bh[2][TN * LDW];
  __shared__ __bf16 Bl[2][TN * LDW];

  const int t    = threadIdx.x;
  const int lane = t & 31;
  const int wave = t >> 5;
  const int m0   = blockIdx.y * TM;
  const int n0   = blockIdx.x * TN;
  const int col  = lane & 15;
  const int half = lane >> 4;

  float4 aReg[4];   // staged global data for one A chunk (128x32 / 256thr)
  float4 bReg[2];   // staged global data for one B chunk (64x32 / 256thr)

  auto gload = [&](int k0) {
#pragma unroll
    for (int i = 0; i < 4; ++i) {
      int slot = t + i * 256;
      int m = slot >> 3, kq = (slot & 7) * 4;
      aReg[i] = *(const float4*)&context[(size_t)(m0 + m) * ND + k0 + kq];
    }
#pragma unroll
    for (int i = 0; i < 2; ++i) {
      int slot = t + i * 256;
      int n = slot >> 3, kq = (slot & 7) * 4;
      bReg[i] = *(const float4*)&Wpre[(size_t)(n0 + n) * ND + k0 + kq];
    }
  };

  auto cstore = [&](int buf) {
#pragma unroll
    for (int i = 0; i < 4; ++i) {
      int slot = t + i * 256;
      int m = slot >> 3, kq = (slot & 7) * 4;
      v4bf h, l;
      split4(aReg[i], h, l);
      *(v4bf*)&Ah[buf][m * LDW + kq] = h;   // 8B stores, 8B aligned
      *(v4bf*)&Al[buf][m * LDW + kq] = l;
    }
#pragma unroll
    for (int i = 0; i < 2; ++i) {
      int slot = t + i * 256;
      int n = slot >> 3, kq = (slot & 7) * 4;
      v4bf h, l;
      split4(bReg[i], h, l);
      *(v4bf*)&Bh[buf][n * LDW + kq] = h;
      *(v4bf*)&Bl[buf][n * LDW + kq] = l;
    }
  };

  v8f acc[4] = {{}, {}, {}, {}};

  // prologue: chunk0 -> buf0, chunk1 staged in regs
  gload(0);
  cstore(0);
  gload(KC);
  __syncthreads();

  // ISA fragment layouts (wave32, bf16):
  //   A 16x32: lane<16 holds K 0-7 and 16-23 (interleaved 8-blocks),
  //            lane>=16 holds K 8-15 and 24-31.
  //   B 32x16: same K order as A (lane<16 K 0-7,16-23; lane>=16 K 8-15,24-31).
  const int rowA = wave * 16 + col;
  const int kA   = 8 * half;    // + {0..7} and +16+{0..7}
  const int kB   = 8 * half;    // + {0..7} and +16+{0..7}

  for (int kc = 0; kc < NCHUNK; ++kc) {
    const int cur = kc & 1;
    if (kc + 1 < NCHUNK) cstore((kc + 1) & 1);   // regs -> next buffer
    if (kc + 2 < NCHUNK) gload((kc + 2) * KC);   // async global for k+2

    v8bf a0 = *(const v8bf*)&Ah[cur][rowA * LDW + kA];
    v8bf a1 = *(const v8bf*)&Ah[cur][rowA * LDW + kA + 16];
    v16bf aHi = __builtin_shufflevector(a0, a1, 0, 1, 2, 3, 4, 5, 6, 7,
                                        8, 9, 10, 11, 12, 13, 14, 15);
    a0 = *(const v8bf*)&Al[cur][rowA * LDW + kA];
    a1 = *(const v8bf*)&Al[cur][rowA * LDW + kA + 16];
    v16bf aLo = __builtin_shufflevector(a0, a1, 0, 1, 2, 3, 4, 5, 6, 7,
                                        8, 9, 10, 11, 12, 13, 14, 15);
#pragma unroll
    for (int nb = 0; nb < 4; ++nb) {
      const int rowB = nb * 16 + col;
      v8bf b0 = *(const v8bf*)&Bh[cur][rowB * LDW + kB];
      v8bf b1 = *(const v8bf*)&Bh[cur][rowB * LDW + kB + 16];
      v16bf bHi = __builtin_shufflevector(b0, b1, 0, 1, 2, 3, 4, 5, 6, 7,
                                          8, 9, 10, 11, 12, 13, 14, 15);
      b0 = *(const v8bf*)&Bl[cur][rowB * LDW + kB];
      b1 = *(const v8bf*)&Bl[cur][rowB * LDW + kB + 16];
      v16bf bLo = __builtin_shufflevector(b0, b1, 0, 1, 2, 3, 4, 5, 6, 7,
                                          8, 9, 10, 11, 12, 13, 14, 15);
      // bf16x3: hi*hi + hi*lo + lo*hi, fp32 accumulate
      acc[nb] = __builtin_amdgcn_wmma_f32_16x16x32_bf16(
          false, aHi, false, bHi, (short)0, acc[nb], false, false);
      acc[nb] = __builtin_amdgcn_wmma_f32_16x16x32_bf16(
          false, aHi, false, bLo, (short)0, acc[nb], false, false);
      acc[nb] = __builtin_amdgcn_wmma_f32_16x16x32_bf16(
          false, aLo, false, bHi, (short)0, acc[nb], false, false);
    }
    __syncthreads();
  }

  // Epilogue. C/D layout: VGPR r -> M=r (lanes 0-15), M=r+8 (lanes 16-31);
  // N = lane%16.
#pragma unroll
  for (int r = 0; r < 8; ++r) {
    int m = m0 + wave * 16 + r + 8 * half;
    int b = m >> 10;               // S == 1024
    float cov = coverage[m];
    float vals[4] = {acc[0][r], acc[1][r], acc[2][r], acc[3][r]};
    float es = 0.f;
#pragma unroll
    for (int nb = 0; nb < 4; ++nb) {
      int n = n0 + nb * 16 + col;
      float v = vals[nb] + bpre[n] + target[b * ND + n] + cov * Wcov[n];
      v = tanhf(v);
      precompute_out[(size_t)m * ND + n] = v;
      es += v * Wv[n];
    }
    es += __shfl_xor(es, 1, 32);
    es += __shfl_xor(es, 2, 32);
    es += __shfl_xor(es, 4, 32);
    es += __shfl_xor(es, 8, 32);
    if (col == 0) atomicAdd(&energy[m], es);
  }
}

// ---------------------------------------------------------------------------
// Row softmax over S, plus coverage_new = coverage + score. One block per b.
// ---------------------------------------------------------------------------
__global__ __launch_bounds__(256)
void softmax_kernel(const float* __restrict__ energy,
                    const float* __restrict__ coverage,
                    float* __restrict__ score,
                    float* __restrict__ covnew) {
  __shared__ float red[256];
  const int b = blockIdx.x, t = threadIdx.x;
  const float* e = energy + (size_t)b * NS;
  float v[4];
  float mx = -INFINITY;
#pragma unroll
  for (int j = 0; j < 4; ++j) { v[j] = e[t + j * 256]; mx = fmaxf(mx, v[j]); }
  red[t] = mx; __syncthreads();
  for (int off = 128; off > 0; off >>= 1) {
    if (t < off) red[t] = fmaxf(red[t], red[t + off]);
    __syncthreads();
  }
  mx = red[0]; __syncthreads();
  float sm = 0.f;
#pragma unroll
  for (int j = 0; j < 4; ++j) { v[j] = expf(v[j] - mx); sm += v[j]; }
  red[t] = sm; __syncthreads();
  for (int off = 128; off > 0; off >>= 1) {
    if (t < off) red[t] += red[t + off];
    __syncthreads();
  }
  const float inv = 1.f / red[0];
#pragma unroll
  for (int j = 0; j < 4; ++j) {
    int s = t + j * 256;
    float p = v[j] * inv;
    score[(size_t)b * NS + s]  = p;
    covnew[(size_t)b * NS + s] = coverage[(size_t)b * NS + s] + p;
  }
}

// ---------------------------------------------------------------------------
// weightedContext[b][a] = sum_s score[b][s] * context[b][s][a]
// score loads wave-uniform; context loads fully coalesced (memory bound).
// ---------------------------------------------------------------------------
__global__ __launch_bounds__(256)
void wctx_kernel(const float* __restrict__ score,
                 const float* __restrict__ context,
                 float* __restrict__ wc) {
  const int b = blockIdx.y;
  const int a = blockIdx.x * 256 + threadIdx.x;
  const float* sc  = score + (size_t)b * NS;
  const float* ctx = context + (size_t)b * NS * ND + a;
  float acc = 0.f;
#pragma unroll 8
  for (int s = 0; s < NS; ++s) acc += sc[s] * ctx[(size_t)s * ND];
  wc[(size_t)b * ND + a] = acc;
}

// ---------------------------------------------------------------------------
extern "C" void kernel_launch(void* const* d_in, const int* in_sizes, int n_in,
                              void* d_out, int out_size, void* d_ws, size_t ws_size,
                              hipStream_t stream) {
  const float* input    = (const float*)d_in[0];  // (B, 1024)
  const float* context  = (const float*)d_in[1];  // (B, S, 1024)
  const float* coverage = (const float*)d_in[2];  // (B, S)
  const float* Wpre     = (const float*)d_in[3];  // (1024, 1024)
  const float* bpre     = (const float*)d_in[4];  // (1024)
  const float* Wq       = (const float*)d_in[5];  // (1024, 1024)
  const float* Wv       = (const float*)d_in[6];  // (1, 1024)
  const float* Wcov     = (const float*)d_in[7];  // (1024, 1)

  float* out        = (float*)d_out;
  float* wc         = out;                 // (B, 1024)
  float* score      = out + MTOT;          // (B, S)
  float* covnew     = out + 2 * MTOT;      // (B, S)
  float* precompute = out + 3 * MTOT;      // (B, S, 1024)

  float* energy = (float*)d_ws;            // (B*S) partial-sum accumulator
  float* target = energy + MTOT;           // (B, 1024)

  (void)hipMemsetAsync(energy, 0, (size_t)MTOT * sizeof(float), stream);

  target_kernel<<<MTOT / 256, 256, 0, stream>>>(input, Wq, target);

  fused_gemm_tanh_energy<<<dim3(ND / TN, MTOT / TM), 256, 0, stream>>>(
      context, Wpre, bpre, target, coverage, Wv, Wcov, precompute, energy);

  softmax_kernel<<<NB, 256, 0, stream>>>(energy, coverage, score, covnew);

  wctx_kernel<<<dim3(ND / 256, NB), 256, 0, stream>>>(score, context, wc);
}
